// Greedy_Base_hGLM_24678882083244
// MI455X (gfx1250) — matrix operand order repaired and checked by reading, not compiled, not run on hardware
//
#include <hip/hip_runtime.h>
#include <hip/hip_bf16.h>
#include <math.h>

// ---------------------------------------------------------------------------
// Problem constants (from reference)
// ---------------------------------------------------------------------------
#define SUB_NO 20
#define T_NO   201
#define E_NO   2000
#define I_NO   500
#define T_DATA 20000
#define K_TOT  2500          // E_NO + I_NO
#define K_PAD  2528          // 79 * 32   (K padded to WMMA f16 K-step)
#define KB_NO  79            // K_PAD / 32
#define N_PAD  128           // 3 variants * 40 channels = 120, padded to 128
#define NT_NO  8             // N_PAD / 16

// Uniform K phases (per 32-wide kb block):
//  kb 0..61   : all k in S_e                (62 blocks)
//  kb 62      : straddles e/i boundary at k=2000
//  kb 63..77  : all k in S_i                (15 blocks)
//  kb 78      : straddles i/zero-pad boundary at k=2500

// d_out layout (floats), reference tuple order:
// V_hard_z, V_soft_z, V_soft_zb, theta, hard_z, soft_z, soft_zb
#define OFF_V      0         // 3 * 20000 (variant-major)
#define OFF_THETA  60000
#define OFF_HARD   110000
#define OFF_SOFT   160000
#define OFF_SOFTB  210000

// workspace layout (bytes)
#define WS_B   0                                   // packed f16 B operand: K_PAD x N_PAD
#define WS_G   ((size_t)K_PAD * N_PAD * 2)         // GEMM out: T_DATA x N_PAD f32
#define WS_F   (WS_G + (size_t)T_DATA * N_PAD * 4) // syn_in: 3 x T_DATA x SUB_NO f32

typedef __attribute__((ext_vector_type(16))) _Float16 v16h;
typedef __attribute__((ext_vector_type(8)))  float    v8f;

// ---------------------------------------------------------------------------
// Packed-B addressing: element (k, n) of the B operand, stored exactly in the
// per-lane order the WMMA B operand expects (CDNA5 ISA 7.12.2 B striping):
//   lane l in [0,16):  N = l,      K = 0..7, 16..23  (half h -> K = h<8 ? h : h+8)
//   lane l in [16,32): N = l - 16, K = 8..15, 24..31 (half h -> K = 8 + (h<8 ? h : h+8))
// Tiles stored as [kb][nb][lane][half] so the GEMM does one contiguous v16h
// load per (kb, nb, lane).
// ---------------------------------------------------------------------------
__device__ __forceinline__ int b_off(int k, int n) {
    int kb = k >> 5, kl = k & 31, nb = n >> 4;
    int lane = ((kl & 8) ? 16 : 0) + (n & 15);
    int h    = (kl & 7) + ((kl & 16) ? 8 : 0);
    return ((kb * NT_NO + nb) * 32 + lane) * 16 + h;
}

// ---------------------------------------------------------------------------
// K1: softmax / Gumbel reparam for 2500 synapse columns + pack B operand.
// ---------------------------------------------------------------------------
__global__ __launch_bounds__(256)
void gumbel_pack_kernel(const float* __restrict__ u, const float* __restrict__ v,
                        const float* __restrict__ Clog,
                        _Float16* __restrict__ Bp, float* __restrict__ out) {
    int col = blockIdx.x * 256 + threadIdx.x;
    if (col >= K_PAD) return;

    // zero this k-row of the packed B operand (covers padding + non-owned n)
    for (int n = 0; n < N_PAD; ++n) Bp[b_off(col, n)] = (_Float16)0.0f;
    if (col >= K_TOT) return;

    const int j = (col < E_NO) ? 0 : 1;   // e-column or i-column

    // softmax over subunits (axis 0)
    float m = -3.0e38f;
    for (int s = 0; s < SUB_NO; ++s) m = fmaxf(m, Clog[s * K_TOT + col]);
    float Z = 0.0f;
    for (int s = 0; s < SUB_NO; ++s) Z += expf(Clog[s * K_TOT + col] - m);

    // pass 2: theta, rebar_z, argmax (first max, matching jnp.argmax)
    float best = -3.0e38f; int arg = 0;
    for (int s = 0; s < SUB_NO; ++s) {
        float th = expf(Clog[s * K_TOT + col] - m) / Z;
        out[OFF_THETA + s * K_TOT + col] = th;
        float rz = logf(th) - logf(-logf(u[s * K_TOT + col]));
        if (rz > best) { best = rz; arg = s; }
    }
    const float lvk = logf(v[arg * K_TOT + col]);   // log(v_k)

    // pass 3: hard/soft/softb + B packing
    for (int s = 0; s < SUB_NO; ++s) {
        float th = expf(Clog[s * K_TOT + col] - m) / Z;
        float rz = logf(th) - logf(-logf(u[s * K_TOT + col]));
        float sz = 1.0f / (1.0f + expf(-2.0f * rz)) + 1e-9f;      // sigmoid(rz/0.5)
        float b  = (s == arg) ? 1.0f : 0.0f;
        float nlv = -logf(v[s * K_TOT + col]);
        float zb  = (s == arg) ? -logf(nlv) : -logf(nlv / th - lvk);
        float szb = 1.0f / (1.0f + expf(-2.0f * zb)) + 1e-9f;

        out[OFF_HARD  + s * K_TOT + col] = b;
        out[OFF_SOFT  + s * K_TOT + col] = sz;
        out[OFF_SOFTB + s * K_TOT + col] = szb;

        // B columns for the fused 3-variant GEMM: n = variant*40 + s*2 + j
        Bp[b_off(col, 0 * 40 + s * 2 + j)] = (_Float16)b;
        Bp[b_off(col, 1 * 40 + s * 2 + j)] = (_Float16)sz;
        Bp[b_off(col, 2 * 40 + s * 2 + j)] = (_Float16)szb;
    }
}

// ---------------------------------------------------------------------------
// K2: fused GEMM  G[T_DATA x 128] = [S_e | S_i] (T x 2500, f32->f16)  x  B
// One wave per 32-row x 128-col panel (two 16-row WMMA M-tiles share every
// B-tile load). A loaded per the CDNA5 16-bit A-operand layout:
//   lane<16:  M = lane,    K halves {0..7, 16..23} of the 32-block
//   lane>=16: M = lane-16, K halves {8..15, 24..31}
// K loop split into wave-uniform phases so the hot path is pure b128 loads.
// ---------------------------------------------------------------------------
template<int BASE>
__device__ __forceinline__ void cvt8(const float* __restrict__ p, v16h& a) {
    const float4* q = reinterpret_cast<const float4*>(p);
    float4 x = q[0], y = q[1];
    a[BASE + 0] = (_Float16)x.x; a[BASE + 1] = (_Float16)x.y;
    a[BASE + 2] = (_Float16)x.z; a[BASE + 3] = (_Float16)x.w;
    a[BASE + 4] = (_Float16)y.x; a[BASE + 5] = (_Float16)y.y;
    a[BASE + 6] = (_Float16)y.z; a[BASE + 7] = (_Float16)y.w;
}

// element-wise fallback for the two boundary kb blocks
__device__ __forceinline__ void loadBoundary(const float* __restrict__ rE,
                                             const float* __restrict__ rI,
                                             int k0, v16h& a) {
#pragma unroll
    for (int h = 0; h < 16; ++h) {
        int k = k0 + ((h < 8) ? h : (h + 8));
        float f = (k < E_NO) ? rE[k] : (k < K_TOT ? rI[k - E_NO] : 0.0f);
        a[h] = (_Float16)f;
    }
}

__device__ __forceinline__ void kstep(const v16h* __restrict__ Bv, int kb, int lane,
                                      const v16h& a0, const v16h& a1,
                                      v8f acc[2][NT_NO]) {
#pragma unroll
    for (int nt = 0; nt < NT_NO; ++nt) {
        v16h b = Bv[(size_t)(kb * NT_NO + nt) * 32 + lane];
        acc[0][nt] = __builtin_amdgcn_wmma_f32_16x16x32_f16(
            false, a0, false, b, (short)0, acc[0][nt], false, false);
        acc[1][nt] = __builtin_amdgcn_wmma_f32_16x16x32_f16(
            false, a1, false, b, (short)0, acc[1][nt], false, false);
    }
}

__global__ __launch_bounds__(128)
void wmma_gemm_kernel(const float* __restrict__ Se, const float* __restrict__ Si,
                      const _Float16* __restrict__ Bp, float* __restrict__ G) {
    const int lane  = threadIdx.x & 31;
    const int wave  = threadIdx.x >> 5;
    const int mpair = blockIdx.x * 4 + wave;           // 625 pairs of M-tiles
    if (mpair >= T_DATA / 32) return;                  // wave-uniform: EXEC all-1s

    const int row0 = mpair * 32 + (lane & 15);
    const int row1 = row0 + 16;
    const int khi  = (lane & 16) ? 8 : 0;
    const v16h* __restrict__ Bv = reinterpret_cast<const v16h*>(Bp);

    const float* __restrict__ rE0 = Se + (size_t)row0 * E_NO;
    const float* __restrict__ rE1 = Se + (size_t)row1 * E_NO;
    const float* __restrict__ rI0 = Si + (size_t)row0 * I_NO;
    const float* __restrict__ rI1 = Si + (size_t)row1 * I_NO;

    v8f zero = {0.f, 0.f, 0.f, 0.f, 0.f, 0.f, 0.f, 0.f};
    v8f acc[2][NT_NO];
#pragma unroll
    for (int nt = 0; nt < NT_NO; ++nt) { acc[0][nt] = zero; acc[1][nt] = zero; }

    // ---- phase 1: kb 0..61, entirely inside S_e (branch-free b128 loads) ----
    for (int kb = 0; kb < 62; ++kb) {
        const int k0 = kb * 32 + khi;
        v16h a0, a1;
        cvt8<0>(rE0 + k0, a0);  cvt8<8>(rE0 + k0 + 16, a0);
        cvt8<0>(rE1 + k0, a1);  cvt8<8>(rE1 + k0 + 16, a1);
        __builtin_prefetch(rE0 + k0 + 256, 0, 1);      // global_prefetch_b8
        __builtin_prefetch(rE1 + k0 + 256, 0, 1);
        kstep(Bv, kb, lane, a0, a1, acc);
    }

    // ---- kb 62: straddles e/i boundary at k=2000 (element-wise) ----
    {
        const int k0 = 62 * 32 + khi;
        v16h a0, a1;
        loadBoundary(rE0, rI0, k0, a0);
        loadBoundary(rE1, rI1, k0, a1);
        kstep(Bv, 62, lane, a0, a1, acc);
    }

    // ---- phase 2: kb 63..77, entirely inside S_i ----
    for (int kb = 63; kb < 78; ++kb) {
        const int k0 = kb * 32 + khi - E_NO;
        v16h a0, a1;
        cvt8<0>(rI0 + k0, a0);  cvt8<8>(rI0 + k0 + 16, a0);
        cvt8<0>(rI1 + k0, a1);  cvt8<8>(rI1 + k0 + 16, a1);
        __builtin_prefetch(rI0 + k0 + 256, 0, 1);
        __builtin_prefetch(rI1 + k0 + 256, 0, 1);
        kstep(Bv, kb, lane, a0, a1, acc);
    }

    // ---- kb 78: straddles i/zero-pad boundary at k=2500 (element-wise) ----
    {
        const int k0 = 78 * 32 + khi;
        v16h a0, a1;
        loadBoundary(rE0, rI0, k0, a0);
        loadBoundary(rE1, rI1, k0, a1);
        kstep(Bv, 78, lane, a0, a1, acc);
    }

    // D layout: VGPR r -> M = r (lanes 0-15) / 8+r (lanes 16-31), N = lane&15
#pragma unroll
    for (int mt = 0; mt < 2; ++mt) {
        const int mBase = mpair * 32 + mt * 16;
#pragma unroll
        for (int nt = 0; nt < NT_NO; ++nt) {
#pragma unroll
            for (int r = 0; r < 8; ++r) {
                int rr = mBase + ((lane & 16) ? r + 8 : r);
                G[(size_t)rr * N_PAD + nt * 16 + (lane & 15)] = acc[mt][nt][r];
            }
        }
    }
}

// ---------------------------------------------------------------------------
// K3: depthwise causal FIR, 201 taps. Kernels depend only on
// (W_syn, Tau_syn, Delta_syn) -> identical across the 3 variants.
// ---------------------------------------------------------------------------
__global__ __launch_bounds__(256)
void fir_kernel(const float* __restrict__ G, const float* __restrict__ Wsyn,
                const float* __restrict__ Tausyn, const float* __restrict__ Dsyn,
                float* __restrict__ F) {
    __shared__ float tapE[T_NO], tapI[T_NO];
    __shared__ float bufE[256 + T_NO - 1], bufI[256 + T_NO - 1];

    const int tid = threadIdx.x;
    const int s   = blockIdx.y;
    const int vv  = blockIdx.z;
    const int t0  = blockIdx.x * 256;

    if (tid < T_NO) {
#pragma unroll
        for (int j = 0; j < 2; ++j) {
            float del = expf(Dsyn[s * 2 + j]);
            float tau = expf(Tausyn[s * 2 + j]);
            float w   = Wsyn[s * 2 + j];
            float tt  = fmaxf((float)tid - del, 0.0f) / tau;
            float tap = tt * expf(-tt) * w;
            if (j == 0) tapE[tid] = tap; else tapI[tid] = tap;
        }
    }
    const int ne = vv * 40 + s * 2;
    for (int i = tid; i < 256 + T_NO - 1; i += 256) {
        int t = t0 - (T_NO - 1) + i;
        bool ok = (t >= 0) && (t < T_DATA);
        bufE[i] = ok ? G[(size_t)t * N_PAD + ne]     : 0.0f;
        bufI[i] = ok ? G[(size_t)t * N_PAD + ne + 1] : 0.0f;
    }
    __syncthreads();

    const int t = t0 + tid;
    if (t < T_DATA) {
        float acc = 0.0f;
#pragma unroll 4
        for (int tau = 0; tau < T_NO; ++tau) {
            int i = tid + (T_NO - 1) - tau;
            acc += tapE[tau] * bufE[i] + tapI[tau] * bufI[i];
        }
        F[((size_t)vv * T_DATA + t) * SUB_NO + s] = acc;
    }
}

// ---------------------------------------------------------------------------
// K4: binary-tree tanh recursion (children of s are 2s+1, 2s+2).
// ---------------------------------------------------------------------------
__global__ __launch_bounds__(256)
void tree_kernel(const float* __restrict__ F, const float* __restrict__ Wsub,
                 const float* __restrict__ Vo, float* __restrict__ out) {
    int id = blockIdx.x * 256 + threadIdx.x;
    if (id >= 3 * T_DATA) return;
    int vv = id / T_DATA, t = id % T_DATA;
    const float* syn = F + ((size_t)vv * T_DATA + t) * SUB_NO;

    float val[SUB_NO];
#pragma unroll
    for (int s = SUB_NO - 1; s >= 0; --s) {
        float x = syn[s];
        if (2 * s + 1 < SUB_NO) x += val[2 * s + 1] * Wsub[2 * s + 1];
        if (2 * s + 2 < SUB_NO) x += val[2 * s + 2] * Wsub[2 * s + 2];
        val[s] = tanhf(x);
    }
    out[OFF_V + (size_t)vv * T_DATA + t] = val[0] * Wsub[0] + Vo[0];
}

// ---------------------------------------------------------------------------
// Launch
// ---------------------------------------------------------------------------
extern "C" void kernel_launch(void* const* d_in, const int* in_sizes, int n_in,
                              void* d_out, int out_size, void* d_ws, size_t ws_size,
                              hipStream_t stream) {
    (void)in_sizes; (void)n_in; (void)out_size; (void)ws_size;
    const float* Se     = (const float*)d_in[0];
    const float* Si     = (const float*)d_in[1];
    const float* u      = (const float*)d_in[2];
    const float* v      = (const float*)d_in[3];
    const float* Wsyn   = (const float*)d_in[4];
    const float* Tausyn = (const float*)d_in[5];
    const float* Dsyn   = (const float*)d_in[6];
    const float* Wsub   = (const float*)d_in[7];
    const float* Vo     = (const float*)d_in[8];
    const float* Clog   = (const float*)d_in[10];
    float* out = (float*)d_out;

    char* ws = (char*)d_ws;
    _Float16* Bp = (_Float16*)(ws + WS_B);
    float*    G  = (float*)(ws + WS_G);
    float*    F  = (float*)(ws + WS_F);

    gumbel_pack_kernel<<<(K_PAD + 255) / 256, 256, 0, stream>>>(u, v, Clog, Bp, out);
    wmma_gemm_kernel<<<(T_DATA / 32 + 3) / 4, 128, 0, stream>>>(Se, Si, Bp, G);
    fir_kernel<<<dim3((T_DATA + 255) / 256, SUB_NO, 3), 256, 0, stream>>>(G, Wsyn, Tausyn, Dsyn, F);
    tree_kernel<<<(3 * T_DATA + 255) / 256, 256, 0, stream>>>(F, Wsub, Vo, out);
}
